// lwm_mamba_30081950941582
// MI455X (gfx1250) — compile-verified, matrix operand order checked by reading, not run on hardware
//
#include <hip/hip_runtime.h>
#include <hip/hip_bf16.h>
#include <math.h>

// ---------------- problem constants (match reference) ----------------
#define DMODEL   128
#define ELEMS    32
#define NLAYERS  12
#define DSTATE   16
#define DCONV    4
#define DINNER   256
#define DTRANK   8
#define DFF      512
#define BATCH    4
#define SEQ      512
#define NMASK    64
#define ROWS     (BATCH*SEQ)     // 2048 token rows
#define DBC_N    (DTRANK + 2*DSTATE)   // 40
#define DBC_LD   48                    // padded row stride for dbc

typedef __attribute__((ext_vector_type(16))) __bf16 v16bf;
typedef __attribute__((ext_vector_type(8)))  float  v8f;

__device__ __forceinline__ float gelu_f(float v) {
    return 0.5f * v * (1.0f + erff(v * 0.70710678118654752f));
}
__device__ __forceinline__ float silu_f(float v) {
    return v * (1.0f / (1.0f + __expf(-v)));
}

// Load one 16-lane-row fragment (16 bf16 K-elements per lane) with optional mask.
// kb0/kb1 select the K-slots this lane owns per the 16-bit A/B WMMA VGPR layout.
__device__ __forceinline__ v16bf load_frag(const float* __restrict__ row,
                                           int k0, int kb0, int kb1, float mask)
{
    v16bf f;
#pragma unroll
    for (int j = 0; j < 8; ++j) {
        f[j]     = (__bf16)(row[k0 + kb0 + j] * mask);
        f[j + 8] = (__bf16)(row[k0 + kb1 + j] * mask);
    }
    return f;
}

// ---------------------------------------------------------------------
// Register-blocked GEMM via v_wmma_f32_16x16x32_bf16.
// One wave computes a 32x32 C macro-tile: 4 accumulators, 2 A-frags + 2
// B-frags per K-step -> 4 WMMAs per K-step (2x fragment reuse vs 16x16).
// C[M,N] = A[M,K] @ W[N,K]^T (+bias)(+act).  M % 32 == 0, K % 32 == 0,
// N arbitrary (branchless column masking keeps EXEC all-ones for WMMA).
// act: 0 = none, 1 = exact GELU
// ---------------------------------------------------------------------
__global__ __launch_bounds__(32)
void gemm_wmma_kernel(const float* __restrict__ A, const float* __restrict__ W,
                      const float* __restrict__ bias, float* __restrict__ C,
                      int M, int N, int K, int ldc, int act)
{
    const int lane = threadIdx.x;
    const int half = lane >> 4;          // 0: lanes 0-15, 1: lanes 16-31
    const int l16  = lane & 15;
    const int tm0  = blockIdx.y * 32;
    const int tn0  = blockIdx.x * 32;

    // K-element slots held by this lane per the 16-bit A/B VGPR layout
    const int kb0 = half ? 8  : 0;
    const int kb1 = half ? 24 : 16;

    const int m0 = tm0 + l16;            // A rows (M is multiple of 32)
    const int m1 = tm0 + 16 + l16;
    int n0 = tn0 + l16;                  // W rows == B columns
    int n1 = tn0 + 16 + l16;
    const float wm0 = (n0 < N) ? 1.0f : 0.0f;
    const float wm1 = (n1 < N) ? 1.0f : 0.0f;
    if (n0 >= N) n0 = N - 1;             // clamp so loads stay in-bounds (branchless)
    if (n1 >= N) n1 = N - 1;

    const float* __restrict__ arow0 = A + (size_t)m0 * K;
    const float* __restrict__ arow1 = A + (size_t)m1 * K;
    const float* __restrict__ wrow0 = W + (size_t)n0 * K;
    const float* __restrict__ wrow1 = W + (size_t)n1 * K;

    v8f c00 = {}, c01 = {}, c10 = {}, c11 = {};
    for (int k0 = 0; k0 < K; k0 += 32) {
        const v16bf a0 = load_frag(arow0, k0, kb0, kb1, 1.0f);
        const v16bf a1 = load_frag(arow1, k0, kb0, kb1, 1.0f);
        const v16bf b0 = load_frag(wrow0, k0, kb0, kb1, wm0);
        const v16bf b1 = load_frag(wrow1, k0, kb0, kb1, wm1);
        __builtin_prefetch(arow0 + k0 + 64, 0, 1);
        __builtin_prefetch(wrow0 + k0 + 64, 0, 1);
        // (neg_a, A, neg_b, B, c_mod, C, reuse_a, reuse_b)
        c00 = __builtin_amdgcn_wmma_f32_16x16x32_bf16(false, a0, false, b0, (short)0, c00, false, false);
        c01 = __builtin_amdgcn_wmma_f32_16x16x32_bf16(false, a0, false, b1, (short)0, c01, false, false);
        c10 = __builtin_amdgcn_wmma_f32_16x16x32_bf16(false, a1, false, b0, (short)0, c10, false, false);
        c11 = __builtin_amdgcn_wmma_f32_16x16x32_bf16(false, a1, false, b1, (short)0, c11, false, false);
    }

    // Epilogue: C layout per accumulator: VGPR i -> row = base + i + 8*half, col = cbase + l16
    const int col0 = tn0 + l16;
    const int col1 = tn0 + 16 + l16;
    const int rbase0 = tm0 + (half ? 8 : 0);
    const int rbase1 = tm0 + 16 + (half ? 8 : 0);

    if (col0 < N) {
        const float bb = bias ? bias[col0] : 0.0f;
#pragma unroll
        for (int i = 0; i < 8; ++i) {
            float v = c00[i] + bb;
            if (act == 1) v = gelu_f(v);
            C[(size_t)(rbase0 + i) * ldc + col0] = v;
            float w = c10[i] + bb;
            if (act == 1) w = gelu_f(w);
            C[(size_t)(rbase1 + i) * ldc + col0] = w;
        }
    }
    if (col1 < N) {
        const float bb = bias ? bias[col1] : 0.0f;
#pragma unroll
        for (int i = 0; i < 8; ++i) {
            float v = c01[i] + bb;
            if (act == 1) v = gelu_f(v);
            C[(size_t)(rbase0 + i) * ldc + col1] = v;
            float w = c11[i] + bb;
            if (act == 1) w = gelu_f(w);
            C[(size_t)(rbase1 + i) * ldc + col1] = w;
        }
    }
}

// ---------------------------------------------------------------------
// LayerNorm over last dim 128, optional residual add. One wave per row,
// 4 elems/lane, shuffle reductions. Safe to run in-place (single read pass).
// ---------------------------------------------------------------------
__global__ void ln_kernel(const float* __restrict__ x, const float* __restrict__ res,
                          const float* __restrict__ g, const float* __restrict__ b,
                          float* __restrict__ out, int rows)
{
    const int lane = threadIdx.x & 31;
    const int wave = threadIdx.x >> 5;
    const int row  = blockIdx.x * (blockDim.x >> 5) + wave;
    if (row >= rows) return;

    const float* xr = x + (size_t)row * DMODEL;
    const float* rr = res ? res + (size_t)row * DMODEL : nullptr;
    float v[4];
#pragma unroll
    for (int i = 0; i < 4; ++i) {
        const int cidx = lane + 32 * i;
        v[i] = xr[cidx] + (rr ? rr[cidx] : 0.0f);
    }
    float s = v[0] + v[1] + v[2] + v[3];
#pragma unroll
    for (int off = 16; off > 0; off >>= 1) s += __shfl_xor(s, off, 32);
    const float mean = s * (1.0f / 128.0f);

    float q = 0.0f;
#pragma unroll
    for (int i = 0; i < 4; ++i) { const float d = v[i] - mean; q += d * d; }
#pragma unroll
    for (int off = 16; off > 0; off >>= 1) q += __shfl_xor(q, off, 32);
    const float rstd = rsqrtf(q * (1.0f / 128.0f) + 1e-5f);

#pragma unroll
    for (int i = 0; i < 4; ++i) {
        const int cidx = lane + 32 * i;
        out[(size_t)row * DMODEL + cidx] = (v[i] - mean) * rstd * g[cidx] + b[cidx];
    }
}

// ---------------------------------------------------------------------
// Embed: input_ids(2048,32) @ proj_w(128,32)^T + proj_b  -> tmp(2048,128)
// ---------------------------------------------------------------------
__global__ void embed_kernel(const float* __restrict__ ids, const float* __restrict__ w,
                             const float* __restrict__ b, float* __restrict__ out)
{
    const int id = blockIdx.x * blockDim.x + threadIdx.x;
    if (id >= ROWS * DMODEL) return;
    const int row = id / DMODEL, d = id % DMODEL;
    float s = b[d];
#pragma unroll
    for (int e = 0; e < ELEMS; ++e) s += ids[row * ELEMS + e] * w[d * ELEMS + e];
    out[id] = s;
}

// ---------------------------------------------------------------------
// Depthwise causal conv (width 4) + bias + SiLU. Reads xi-half of xz.
// ---------------------------------------------------------------------
__global__ void conv_silu_kernel(const float* __restrict__ xz, const float* __restrict__ cw,
                                 const float* __restrict__ cb, float* __restrict__ xi)
{
    const int id = blockIdx.x * blockDim.x + threadIdx.x;
    if (id >= BATCH * SEQ * DINNER) return;
    const int d = id % DINNER;
    const int t = (id / DINNER) % SEQ;
    const int b = id / (DINNER * SEQ);
    float acc = cb[d];
#pragma unroll
    for (int j = 0; j < DCONV; ++j) {
        const int tt = t - (DCONV - 1) + j;
        if (tt >= 0) acc += cw[d * DCONV + j] * xz[((size_t)(b * SEQ + tt)) * (2 * DINNER) + d];
    }
    xi[id] = silu_f(acc);
}

// ---------------------------------------------------------------------
// dt = softplus(dbc[:, :8] @ dtw(256,8)^T + dtb)
// ---------------------------------------------------------------------
__global__ void dt_kernel(const float* __restrict__ dbc, const float* __restrict__ dtw,
                          const float* __restrict__ dtb, float* __restrict__ dt)
{
    const int id = blockIdx.x * blockDim.x + threadIdx.x;
    if (id >= ROWS * DINNER) return;
    const int row = id / DINNER, d = id % DINNER;
    const float* db = dbc + (size_t)row * DBC_LD;
    float s = dtb[d];
#pragma unroll
    for (int r = 0; r < DTRANK; ++r) s += db[r] * dtw[d * DTRANK + r];
    dt[id] = (s > 20.0f) ? s : log1pf(__expf(s));
}

// ---------------------------------------------------------------------
// Selective scan: per (b,d) recurrence over L=512 with 16 states in VGPRs.
// Fuses +xi*D and the SiLU(z) gate. y(2048,256).
// ---------------------------------------------------------------------
__global__ void scan_kernel(const float* __restrict__ dt, const float* __restrict__ xi,
                            const float* __restrict__ dbc, const float* __restrict__ xz,
                            const float* __restrict__ A_log, const float* __restrict__ Dp,
                            float* __restrict__ y)
{
    const int id = blockIdx.x * blockDim.x + threadIdx.x;
    if (id >= BATCH * DINNER) return;
    const int b = id / DINNER, d = id % DINNER;

    float A[DSTATE], h[DSTATE];
#pragma unroll
    for (int n = 0; n < DSTATE; ++n) { A[n] = -__expf(A_log[d * DSTATE + n]); h[n] = 0.0f; }
    const float Dv = Dp[d];

    for (int t = 0; t < SEQ; ++t) {
        const int r = b * SEQ + t;
        const float dtv = dt[(size_t)r * DINNER + d];
        const float xv  = xi[(size_t)r * DINNER + d];
        const float zv  = xz[(size_t)r * (2 * DINNER) + DINNER + d];
        const float* db = dbc + (size_t)r * DBC_LD;
        float acc = 0.0f;
#pragma unroll
        for (int n = 0; n < DSTATE; ++n) {
            const float Bn = db[DTRANK + n];
            const float Cn = db[DTRANK + DSTATE + n];
            h[n] = __expf(dtv * A[n]) * h[n] + dtv * Bn * xv;
            acc += h[n] * Cn;
        }
        y[(size_t)r * DINNER + d] = (acc + xv * Dv) * silu_f(zv);
    }
}

// ---------------------------------------------------------------------
// Gather masked positions: hm[b,m,:] = cur[b, pos[b,m], :]
// ---------------------------------------------------------------------
__global__ void gather_kernel(const float* __restrict__ cur, const int* __restrict__ pos,
                              float* __restrict__ hm)
{
    const int id = blockIdx.x * blockDim.x + threadIdx.x;
    if (id >= BATCH * NMASK * DMODEL) return;
    const int c = id % DMODEL;
    const int bm = id / DMODEL;
    const int b = bm / NMASK;
    int p = pos[bm];
    if (p < 0) p = 0;
    if (p >= SEQ) p = SEQ - 1;
    hm[id] = cur[((size_t)(b * SEQ + p)) * DMODEL + c];
}

// ---------------------------------------------------------------------
static inline void launch_gemm(const float* A, const float* W, const float* bias,
                               float* C, int M, int N, int K, int ldc, int act,
                               hipStream_t s)
{
    dim3 grid((N + 31) / 32, (M + 31) / 32);
    gemm_wmma_kernel<<<grid, 32, 0, s>>>(A, W, bias, C, M, N, K, ldc, act);
}

extern "C" void kernel_launch(void* const* d_in, const int* in_sizes, int n_in,
                              void* d_out, int out_size, void* d_ws, size_t ws_size,
                              hipStream_t stream)
{
    (void)in_sizes; (void)n_in; (void)out_size; (void)ws_size;

    // ---- inputs (setup_inputs order) ----
    const float* input_ids = (const float*)d_in[0];
    const int*   masked    = (const int*)  d_in[1];
    const float* proj_w    = (const float*)d_in[2];
    const float* proj_b    = (const float*)d_in[3];
    const float* inorm_g   = (const float*)d_in[4];
    const float* inorm_b   = (const float*)d_in[5];
    const float* in_proj_w = (const float*)d_in[6];
    const float* conv_w    = (const float*)d_in[7];
    const float* conv_b    = (const float*)d_in[8];
    const float* x_proj_w  = (const float*)d_in[9];
    const float* dt_proj_w = (const float*)d_in[10];
    const float* dt_proj_b = (const float*)d_in[11];
    const float* A_log     = (const float*)d_in[12];
    const float* D_param   = (const float*)d_in[13];
    const float* out_proj_w= (const float*)d_in[14];
    const float* ln1_g = (const float*)d_in[15];
    const float* ln1_b = (const float*)d_in[16];
    const float* ln2_g = (const float*)d_in[17];
    const float* ln2_b = (const float*)d_in[18];
    const float* ffn_w1 = (const float*)d_in[19];
    const float* ffn_b1 = (const float*)d_in[20];
    const float* ffn_w2 = (const float*)d_in[21];
    const float* ffn_b2 = (const float*)d_in[22];
    const float* ln3_g = (const float*)d_in[23];
    const float* ln3_b = (const float*)d_in[24];
    const float* lin_w = (const float*)d_in[25];
    const float* lin_b = (const float*)d_in[26];
    const float* norm_g = (const float*)d_in[27];
    const float* norm_b = (const float*)d_in[28];
    const float* dec_w = (const float*)d_in[29];
    const float* dec_b = (const float*)d_in[30];

    // ---- workspace layout (floats) ----
    float* ws   = (float*)d_ws;
    float* cur  = ws;                    // (2048,128) residual stream
    float* buf1 = cur  + ROWS * DMODEL;  // (2048,128) LN temp / h
    float* mix  = buf1 + ROWS * DMODEL;  // (2048,128) mixer / ffn2 out
    float* xz   = mix  + ROWS * DMODEL;  // (2048,512)
    float* xi   = xz   + ROWS * 2 * DINNER; // (2048,256)
    float* dbc  = xi   + ROWS * DINNER;  // (2048,48 padded; 40 valid)
    float* dtb_ = dbc  + ROWS * DBC_LD;  // (2048,256)
    float* yv   = dtb_ + ROWS * DINNER;  // (2048,256)
    float* ffn1 = yv   + ROWS * DINNER;  // (2048,512)
    float* hm   = ffn1 + ROWS * DFF;     // (256,128)
    float* hm2  = hm   + BATCH * NMASK * DMODEL; // (256,128)

    float* logits = (float*)d_out;                       // (256,32)
    float* out_seq = (float*)d_out + BATCH * NMASK * ELEMS; // (2048,128)

    const int LN_BLOCK = 256;                 // 8 waves = 8 rows / block
    const int LN_GRID  = (ROWS + 7) / 8;
    const int LN_GRID_M = (BATCH * NMASK + 7) / 8;

    // ---- embedding + input LN ----
    embed_kernel<<<(ROWS * DMODEL + 255) / 256, 256, 0, stream>>>(input_ids, proj_w, proj_b, buf1);
    ln_kernel<<<LN_GRID, LN_BLOCK, 0, stream>>>(buf1, nullptr, inorm_g, inorm_b, cur, ROWS);

    // ---- 12 layers ----
    for (int i = 0; i < NLAYERS; ++i) {
        // h = LN1(out)
        ln_kernel<<<LN_GRID, LN_BLOCK, 0, stream>>>(cur, nullptr, ln1_g + i * DMODEL,
                                                    ln1_b + i * DMODEL, buf1, ROWS);
        // xz = h @ in_proj_w^T          (2048 x 512, K=128)
        launch_gemm(buf1, in_proj_w + (size_t)i * 2 * DINNER * DMODEL, nullptr,
                    xz, ROWS, 2 * DINNER, DMODEL, 2 * DINNER, 0, stream);
        // xi = silu(conv(xz[:, :256]) + cb)
        conv_silu_kernel<<<(BATCH * SEQ * DINNER + 255) / 256, 256, 0, stream>>>(
            xz, conv_w + (size_t)i * DINNER * DCONV, conv_b + (size_t)i * DINNER, xi);
        // dbc = xi @ x_proj_w^T         (2048 x 40, K=256, ldc=48)
        launch_gemm(xi, x_proj_w + (size_t)i * DBC_N * DINNER, nullptr,
                    dbc, ROWS, DBC_N, DINNER, DBC_LD, 0, stream);
        // dt = softplus(dbc[:, :8] @ dtw^T + dtb)
        dt_kernel<<<(ROWS * DINNER + 255) / 256, 256, 0, stream>>>(
            dbc, dt_proj_w + (size_t)i * DINNER * DTRANK, dt_proj_b + (size_t)i * DINNER, dtb_);
        // selective scan + D skip + SiLU(z) gate
        scan_kernel<<<(BATCH * DINNER + 127) / 128, 128, 0, stream>>>(
            dtb_, xi, dbc, xz, A_log + (size_t)i * DINNER * DSTATE,
            D_param + (size_t)i * DINNER, yv);
        // mix = y @ out_proj_w^T        (2048 x 128, K=256)
        launch_gemm(yv, out_proj_w + (size_t)i * DMODEL * DINNER, nullptr,
                    mix, ROWS, DMODEL, DINNER, DMODEL, 0, stream);
        // out = LN2(res + mix)  (in-place on cur is safe: one read pass per wave-row)
        ln_kernel<<<LN_GRID, LN_BLOCK, 0, stream>>>(mix, cur, ln2_g + i * DMODEL,
                                                    ln2_b + i * DMODEL, cur, ROWS);
        // ffn1 = gelu(out @ w1^T + b1)  (2048 x 512, K=128)
        launch_gemm(cur, ffn_w1 + (size_t)i * DFF * DMODEL, ffn_b1 + (size_t)i * DFF,
                    ffn1, ROWS, DFF, DMODEL, DFF, 1, stream);
        // mix = ffn1 @ w2^T + b2        (2048 x 128, K=512)
        launch_gemm(ffn1, ffn_w2 + (size_t)i * DMODEL * DFF, ffn_b2 + (size_t)i * DMODEL,
                    mix, ROWS, DMODEL, DFF, DMODEL, 0, stream);
        // out = LN3(res + ffn)
        ln_kernel<<<LN_GRID, LN_BLOCK, 0, stream>>>(mix, cur, ln3_g + i * DMODEL,
                                                    ln3_b + i * DMODEL, cur, ROWS);
    }

    // ---- MLM head ----
    gather_kernel<<<(BATCH * NMASK * DMODEL + 255) / 256, 256, 0, stream>>>(cur, masked, hm);
    // hm2 = gelu(hm @ lin_w^T + lin_b)   (256 x 128, K=128)
    launch_gemm(hm, lin_w, lin_b, hm2, BATCH * NMASK, DMODEL, DMODEL, DMODEL, 1, stream);
    // hm2 = LN(hm2)
    ln_kernel<<<LN_GRID_M, LN_BLOCK, 0, stream>>>(hm2, nullptr, norm_g, norm_b, hm2, BATCH * NMASK);
    // logits = hm2 @ dec_w^T + dec_b     (256 x 32, K=128)
    launch_gemm(hm2, dec_w, dec_b, logits, BATCH * NMASK, ELEMS, DMODEL, ELEMS, 0, stream);

    // second tuple element: the full sequence output
    hipMemcpyAsync(out_seq, cur, (size_t)ROWS * DMODEL * sizeof(float),
                   hipMemcpyDeviceToDevice, stream);
}
